// AreaPolyLoss_4071628996809
// MI455X (gfx1250) — compile-verified, matrix operand order checked by reading, not run on hardware
//
#include <hip/hip_runtime.h>
#include <hip/hip_bf16.h>
#include <hip/hip_fp16.h>

// Problem constants from the reference
#define B_ 8
#define K_ 32
#define C_ 32
#define H_ 128
#define W_ 128
#define V_ 512            // K_ * C_ / 2 vertices (= edges) per batch

typedef __attribute__((ext_vector_type(16))) _Float16 v16h;
typedef __attribute__((ext_vector_type(8)))  float    v8f;

// Workspace layout (floats):
//   [0            .. B_*V_-1 ]  vx[b][v]
//   [B_*V_        .. 2*B_*V_-1] vy[b][v]
//   [2*B_*V_      .. 2*B_*V_+B_-1] per-batch SSE accumulators
#define WS_VX   0
#define WS_VY   (B_*V_)
#define WS_SSE  (2*B_*V_)

__global__ void app_zero_sse(float* __restrict__ ws) {
    int i = blockIdx.x * blockDim.x + threadIdx.x;
    if (i < B_) ws[WS_SSE + i] = 0.0f;
}

// Gather pred from output at ind, add centers, trunc -> vertex arrays.
__global__ void app_vertex_kernel(const float* __restrict__ output,
                                  const int*   __restrict__ ind,
                                  const float* __restrict__ centers,
                                  float* __restrict__ ws) {
    int i = blockIdx.x * blockDim.x + threadIdx.x;   // 0 .. B_*V_-1
    if (i >= B_ * V_) return;
    int b = i / V_;
    int v = i - b * V_;
    int k = v >> 4;           // keypoint
    int j = v & 15;           // coordinate pair index within the C channels
    int idx = ind[b * K_ + k];
    const float* ob = output + (size_t)b * C_ * (H_ * W_);
    float px = ob[(size_t)(2 * j)     * (H_ * W_) + idx];
    float py = ob[(size_t)(2 * j + 1) * (H_ * W_) + idx];
    float cx = centers[(b * K_ + k) * 2 + 0];
    float cy = centers[(b * K_ + k) * 2 + 1];
    ws[WS_VX + i] = truncf(px + cx);
    ws[WS_VY + i] = truncf(py + cy);
}

// One workgroup (256 threads = 8 waves) per (batch, row).
// Phase 0: per-edge scanline x-intersections into LDS.
// Phase 1: per wave, 16 pixels; crossing counts accumulated with
//          v_wmma_f32_16x16x32_f16 (A = 0/1 indicators, B = ones).
// Phase 2: parity -> mask value -> squared error vs target, block reduce,
//          atomicAdd into the per-batch SSE.
__global__ void app_raster_kernel(const float* __restrict__ target,
                                  float* __restrict__ ws) {
    __shared__ float xe[V_];     // x-intersection (or -inf) per edge, this row
    __shared__ float cnt[W_];    // crossing count per pixel
    __shared__ float red[256];   // reduction scratch

    const int bid = blockIdx.x;
    const int b   = bid / H_;
    const int y   = bid - b * H_;
    const float py = (float)y;
    const int tid = threadIdx.x;

    const float* vx = ws + WS_VX + b * V_;
    const float* vy = ws + WS_VY + b * V_;
    const float* trow = target + (size_t)(b * H_ + y) * W_;
    __builtin_prefetch(trow, 0, 3);          // global_prefetch_b8

    // Phase 0: edge x-intersections with this scanline
    for (int e = tid; e < V_; e += 256) {
        float x1 = vx[e], y1 = vy[e];
        int   e2 = (e + 1) & (V_ - 1);
        float x2 = vx[e2], y2 = vy[e2];
        bool straddle = (y1 <= py) != (y2 <= py);
        float xi = -1e30f;                   // px < xi never true if no straddle
        if (straddle) {                      // straddle implies y2 != y1
            float t = (py - y1) / (y2 - y1);
            xi = x1 + t * (x2 - x1);
        }
        xe[e] = xi;
    }
    __syncthreads();

    // Phase 1: WMMA crossing-count accumulation.
    const int lane = tid & 31;
    const int wave = tid >> 5;
    const int m    = lane & 15;                    // A-matrix row = pixel in tile
    const float pxf = (float)(wave * 16 + m);      // pixel x coordinate
    const int hi   = (lane >= 16) ? 8 : 0;         // K-offset for upper half-wave

    v8f acc = {};
    v16h ones;
    #pragma unroll
    for (int t = 0; t < 16; ++t) ones[t] = (_Float16)1.0f;

    for (int c0 = 0; c0 < V_; c0 += 32) {          // 16 chunks of 32 edges
        v16h a;
        #pragma unroll
        for (int r = 0; r < 8; ++r) {
            // 16-bit A 16x32 layout: VGPR r<4 -> K = 2r,2r+1 (+hi)
            //                        VGPR r>=4 -> K = 16+2(r-4),+1 (+hi)
            int kb = (r < 4) ? (2 * r + hi) : (16 + 2 * (r - 4) + hi);
            a[2 * r + 0] = (pxf < xe[c0 + kb + 0]) ? (_Float16)1.0f : (_Float16)0.0f;
            a[2 * r + 1] = (pxf < xe[c0 + kb + 1]) ? (_Float16)1.0f : (_Float16)0.0f;
        }
        acc = __builtin_amdgcn_wmma_f32_16x16x32_f16(
            /*neg_a=*/false, a, /*neg_b=*/false, ones,
            /*c_mod=*/(short)0, acc, /*reuse_a=*/false, /*reuse_b=*/false);
    }

    // C/D layout: lane n<16, VGPR r -> (M=r, N=n); lane n>=16 -> (M=r+8, N=n-16).
    // Column N=0 lives in lanes 0 (M=0..7) and 16 (M=8..15).
    if (lane == 0 || lane == 16) {
        int base = wave * 16 + ((lane == 16) ? 8 : 0);
        #pragma unroll
        for (int r = 0; r < 8; ++r) cnt[base + r] = acc[r];
    }
    __syncthreads();

    // Phase 2: parity -> 0/255 mask, squared error vs target
    float partial = 0.0f;
    if (tid < W_) {
        int c = (int)(cnt[tid] + 0.5f);
        float pred = (c & 1) ? 255.0f : 0.0f;
        float d = pred - trow[tid];
        partial = d * d;
    }
    red[tid] = partial;
    __syncthreads();
    for (int s = 128; s > 0; s >>= 1) {
        if (tid < s) red[tid] += red[tid + s];
        __syncthreads();
    }
    if (tid == 0) atomicAdd(&ws[WS_SSE + b], red[0]);
}

__global__ void app_final_kernel(const float* __restrict__ mask,
                                 const float* __restrict__ ws,
                                 float* __restrict__ out) {
    __shared__ float red[256];
    int tid = threadIdx.x;
    red[tid] = (tid < B_ * K_) ? mask[tid] : 0.0f;   // B_*K_ == 256
    __syncthreads();
    for (int s = 128; s > 0; s >>= 1) {
        if (tid < s) red[tid] += red[tid + s];
        __syncthreads();
    }
    if (tid == 0) {
        float msum = red[0];
        float total = 0.0f;
        #pragma unroll
        for (int b = 0; b < B_; ++b)
            total += ws[WS_SSE + b] * (1.0f / (float)(H_ * W_));
        out[0] = total / (msum * (float)C_ + 0.0001f);
    }
}

extern "C" void kernel_launch(void* const* d_in, const int* in_sizes, int n_in,
                              void* d_out, int out_size, void* d_ws, size_t ws_size,
                              hipStream_t stream) {
    const float* output  = (const float*)d_in[0];   // (B,C,H,W) f32
    const float* mask    = (const float*)d_in[1];   // (B,K)     f32
    const int*   ind     = (const int*)  d_in[2];   // (B,K)     i32
    const float* target  = (const float*)d_in[3];   // (B,H,W)   f32
    const float* centers = (const float*)d_in[4];   // (B,K,2)   f32
    float* out = (float*)d_out;
    float* ws  = (float*)d_ws;

    app_zero_sse<<<1, 64, 0, stream>>>(ws);
    app_vertex_kernel<<<(B_ * V_ + 255) / 256, 256, 0, stream>>>(output, ind, centers, ws);
    app_raster_kernel<<<B_ * H_, 256, 0, stream>>>(target, ws);
    app_final_kernel<<<1, 256, 0, stream>>>(mask, ws, out);
}
